// CNNMoETransformer_85590108275223
// MI455X (gfx1250) — compile-verified, matrix operand order checked by reading, not run on hardware
//
#include <hip/hip_runtime.h>
#include <hip/hip_bf16.h>

// ---------------------------------------------------------------------------
// CNN -> MoE -> LayerNorm -> classifier for MI455X (gfx1250, wave32, WMMA).
// Matrix work: v_wmma_f32_16x16x32_bf16 (bf16 in, f32 accumulate).
// GEMM A-tiles are staged to LDS with async global->LDS DMA (ASYNCcnt),
// double-buffered, and shared by all 8 waves of a block.
// ---------------------------------------------------------------------------

typedef __attribute__((ext_vector_type(16))) __bf16 v16bf;
typedef __attribute__((ext_vector_type(8)))  float  v8f;
typedef int async_i64 __attribute__((vector_size(8)));   // b64 async payload type

#define TPB 256   // 8 waves per block (wave32)

#define ASG __attribute__((address_space(1)))
#define ASL __attribute__((address_space(3)))

#if __has_builtin(__builtin_amdgcn_global_load_async_to_lds_b64)
#define HAVE_ASYNC_LDS 1
#else
#define HAVE_ASYNC_LDS 0
#endif

static __device__ __forceinline__ void wait_async_lds() {
#if HAVE_ASYNC_LDS
#if __has_builtin(__builtin_amdgcn_s_wait_asynccnt)
    __builtin_amdgcn_s_wait_asynccnt(0);
#else
    asm volatile("s_wait_asynccnt 0x0" ::: "memory");
#endif
#endif
}

// Cooperatively stage a 16x32 f32 A-slab (rows m0..m0+15, cols kk..kk+31)
// into LDS: 512 floats / 256 threads = 2 floats (8 bytes) per thread.
static __device__ __forceinline__ void stage_a_slab(
    const float* __restrict__ A, int m0, int K, int kk,
    float* __restrict__ dst, int t)
{
    int lin  = t * 2;
    int srow = lin >> 5;
    int scol = lin & 31;
    const float* g = A + (size_t)(m0 + srow) * K + kk + scol;
    float* l = dst + srow * 32 + scol;
#if HAVE_ASYNC_LDS
    __builtin_amdgcn_global_load_async_to_lds_b64(
        (ASG async_i64*)g, (ASL async_i64*)l, 0, 0);
#else
    l[0] = g[0];
    l[1] = g[1];
#endif
}

// Shared WMMA mainloop: acc = A[m0:m0+16, :K] * Bt[n-col rows]^T.
// A is staged via LDS double buffer; B fragments come straight from global
// (per-wave columns, bf16 pairs contiguous in the pre-transposed layout).
static __device__ __forceinline__ v8f wmma_mainloop(
    const float* __restrict__ A, const __bf16* __restrict__ brow,
    float Atile[2][512], int m0, int K, bool active, int t, int lane)
{
    int hi = lane >> 4, r = lane & 15;
    v8f acc = {};
    stage_a_slab(A, m0, K, 0, Atile[0], t);
    int buf = 0;
    for (int k0 = 0; k0 < K; k0 += 32) {
        wait_async_lds();            // my slab DMA complete
        __syncthreads();             // everyone's slab visible; prev reads done
        if (k0 + 32 < K)
            stage_a_slab(A, m0, K, k0 + 32, Atile[buf ^ 1], t);  // overlap DMA
        if (active) {
            const float* arow = Atile[buf] + r * 32;             // LDS reads
            __builtin_prefetch(brow + k0 + 512, 0, 1);           // global_prefetch_b8
            v16bf af, bfv;
#pragma unroll
            for (int v = 0; v < 8; ++v) {
                int kbl = (((v & 4) ? 16 + 2 * (v & 3) : 2 * v) + 8 * hi);
                float2 a2 = *(const float2*)(arow + kbl);
                af[2 * v]     = (__bf16)a2.x;
                af[2 * v + 1] = (__bf16)a2.y;
                bfv[2 * v]     = brow[k0 + kbl];
                bfv[2 * v + 1] = brow[k0 + kbl + 1];
            }
            acc = __builtin_amdgcn_wmma_f32_16x16x32_bf16(
                      false, af, false, bfv, (short)0, acc, false, false);
        }
        buf ^= 1;
    }
    return acc;
}

// ---------------------------------------------------------------------------
// Weight conversion kernels (f32 -> bf16, with K-padding / transpose)
// ---------------------------------------------------------------------------

// in [M, Kreal] row-major f32  ->  out [M, Kpad] row-major bf16 (zero padded)
__global__ __launch_bounds__(TPB) void cvt_pad_bf16(
    const float* __restrict__ in, unsigned short* __restrict__ out,
    int Kreal, int Kpad)
{
    int k = blockIdx.x * TPB + threadIdx.x;
    int m = blockIdx.y;
    if (k >= Kpad) return;
    __bf16* o = (__bf16*)out;
    float v = (k < Kreal) ? in[(size_t)m * Kreal + k] : 0.0f;
    o[(size_t)m * Kpad + k] = (__bf16)v;
}

// in [K, N] row-major f32  ->  out [Npad, K] row-major bf16 (B^T layout,
// rows n >= N zero-filled).  grid.y = Npad.
__global__ __launch_bounds__(TPB) void cvt_transpose_bf16(
    const float* __restrict__ in, unsigned short* __restrict__ out,
    int K, int N)
{
    int k = blockIdx.x * TPB + threadIdx.x;
    int n = blockIdx.y;
    if (k >= K) return;
    __bf16* o = (__bf16*)out;
    float v = (n < N) ? in[(size_t)k * N + n] : 0.0f;
    o[(size_t)n * K + k] = (__bf16)v;
}

__global__ __launch_bounds__(TPB) void zero_f32(float* __restrict__ p, int n)
{
    int i = blockIdx.x * TPB + threadIdx.x;
    if (i < n) p[i] = 0.0f;
}

// ---------------------------------------------------------------------------
// Generic GEMM:  C[M, Nstore] (+bias, opt relu) = A[M,K](f32) * Bt[Npad,K](bf16)^T
// One 16x16 C tile per wave; 8 waves share one m-tile (A staged once in LDS).
// ---------------------------------------------------------------------------
__global__ __launch_bounds__(TPB) void gemm_bf16_wmma(
    const float* __restrict__ A, const unsigned short* __restrict__ Bt_,
    const float* __restrict__ bias, float* __restrict__ C,
    int M, int Npad, int K, int ldc, int Nstore, int relu)
{
    __shared__ float Atile[2][512];
    const __bf16* Bt = (const __bf16*)Bt_;
    int t    = threadIdx.x;
    int lane = t & 31;
    int wv   = t >> 5;
    int n0 = (blockIdx.x * 8 + wv) * 16;
    int m0 = blockIdx.y * 16;
    bool active = (n0 < Npad);                  // wave-uniform
    int hi = lane >> 4, r = lane & 15;
    const __bf16* brow = Bt + (size_t)(active ? n0 + r : 0) * K;

    v8f acc = wmma_mainloop(A, brow, Atile, m0, K, active, t, lane);

    int cn = n0 + r;
    if (active && cn < Nstore) {
        float bz = bias ? bias[cn] : 0.0f;
#pragma unroll
        for (int v = 0; v < 8; ++v) {
            int cm = m0 + v + 8 * hi;
            float val = acc[v] + bz;
            if (relu) val = fmaxf(val, 0.0f);
            C[(size_t)cm * ldc + cn] = val;
        }
    }
}

// ---------------------------------------------------------------------------
// Expert second GEMM with gate-weighted accumulate:
//   moe[b,:] += gates[b,e] * (relu_h1[b,:] @ W2[e] + b2[e])
// K = 1024, N = 256.  Serialized across expert launches -> deterministic.
// ---------------------------------------------------------------------------
__global__ __launch_bounds__(TPB) void expert2_wmma(
    const float* __restrict__ H1, const unsigned short* __restrict__ W2t_,
    const float* __restrict__ b2, const float* __restrict__ gates,
    int e, float* __restrict__ moe)
{
    __shared__ float Atile[2][512];
    const __bf16* Bt = (const __bf16*)W2t_;
    const int K = 1024;
    int t    = threadIdx.x;
    int lane = t & 31;
    int wv   = t >> 5;
    int n0 = (blockIdx.x * 8 + wv) * 16;        // grid.x = 2  -> n0 in [0,240]
    int m0 = blockIdx.y * 16;
    int hi = lane >> 4, r = lane & 15;
    const __bf16* brow = Bt + (size_t)(n0 + r) * K;

    v8f acc = wmma_mainloop(H1, brow, Atile, m0, K, true, t, lane);

    int cn = n0 + r;
    float bz = b2[cn];
#pragma unroll
    for (int v = 0; v < 8; ++v) {
        int cm = m0 + v + 8 * hi;
        float g = gates[(size_t)cm * 16 + e];
        moe[(size_t)cm * 256 + cn] += g * (acc[v] + bz);
    }
}

// ---------------------------------------------------------------------------
// Implicit-GEMM conv 3x3 SAME stride 1, fused bias + relu.
//   M = Cout (weights, bf16 [Cout,Kpad]),  N = B*H*W pixels,  K = Cin*9.
// Images are square with power-of-two side (1<<logW).  Output NCHW f32.
// ---------------------------------------------------------------------------
__global__ __launch_bounds__(TPB) void conv_wmma(
    const float* __restrict__ X, const unsigned short* __restrict__ Wb_,
    const float* __restrict__ bias, float* __restrict__ Y,
    int Cin, int Cout, int logW, int Kreal, int Kpad)
{
    const __bf16* Wb = (const __bf16*)Wb_;
    int lane = threadIdx.x & 31;
    int wv   = threadIdx.x >> 5;
    int hi = lane >> 4, r = lane & 15;
    int Wimg  = 1 << logW;
    int logHW = logW << 1;
    int ntile = blockIdx.x * 8 + wv;
    int m0    = blockIdx.y * 16;
    int n    = ntile * 16 + r;                  // pixel for this lane
    int bimg = n >> logHW;
    int p    = n & ((1 << logHW) - 1);
    int yy   = p >> logW;
    int xx   = p & (Wimg - 1);
    const __bf16* wrow = Wb + (size_t)(m0 + r) * Kpad;

    v8f acc = {};
    for (int k0 = 0; k0 < Kpad; k0 += 32) {
        v16bf af, bfv;
#pragma unroll
        for (int v = 0; v < 8; ++v) {
            int kb = (((v & 4) ? 16 + 2 * (v & 3) : 2 * v) + 8 * hi) + k0;
            af[2 * v]     = wrow[kb];
            af[2 * v + 1] = wrow[kb + 1];
#pragma unroll
            for (int j = 0; j < 2; ++j) {
                int k   = kb + j;
                int cin = k / 9;                // constant divisor -> mul-shift
                int rem = k - cin * 9;
                int dy  = rem / 3;
                int dx  = rem - dy * 3;
                int iy  = yy + dy - 1;
                int ix  = xx + dx - 1;
                float val = 0.0f;
                bool ok = (k < Kreal) & (iy >= 0) & (iy < Wimg) &
                          (ix >= 0) & (ix < Wimg);
                if (ok) {
                    size_t idx = ((((size_t)(bimg * Cin + cin) << logW) + iy)
                                  << logW) + ix;
                    val = X[idx];
                }
                bfv[2 * v + j] = (__bf16)val;
            }
        }
        acc = __builtin_amdgcn_wmma_f32_16x16x32_bf16(
                  false, af, false, bfv, (short)0, acc, false, false);
    }
    // store NCHW, fused bias + relu (relu commutes with maxpool)
#pragma unroll
    for (int v = 0; v < 8; ++v) {
        int cm = m0 + v + 8 * hi;               // output channel
        float val = fmaxf(acc[v] + bias[cm], 0.0f);
        size_t idx = ((((size_t)(bimg * Cout + cm) << logW) + yy) << logW) + xx;
        Y[idx] = val;
    }
}

// ---------------------------------------------------------------------------
// 2x2 max pool, NCHW, power-of-two dims (logC/logHo/logWo of the OUTPUT).
// ---------------------------------------------------------------------------
__global__ __launch_bounds__(TPB) void maxpool2(
    const float* __restrict__ in, float* __restrict__ out,
    int logC, int logHo, int logWo, int total)
{
    int idx = blockIdx.x * TPB + threadIdx.x;
    if (idx >= total) return;
    int xo = idx & ((1 << logWo) - 1);
    int yo = (idx >> logWo) & ((1 << logHo) - 1);
    int c  = (idx >> (logWo + logHo)) & ((1 << logC) - 1);
    int b  = idx >> (logWo + logHo + logC);
    int plane = (b << logC) | c;
    size_t base = ((size_t)plane << (logHo + logWo + 2)) +
                  ((size_t)(yo << 1) << (logWo + 1)) + (xo << 1);
    int Wi = 1 << (logWo + 1);
    float m0 = fmaxf(in[base],      in[base + 1]);
    float m1 = fmaxf(in[base + Wi], in[base + Wi + 1]);
    out[idx] = fmaxf(m0, m1);
}

// ---------------------------------------------------------------------------
// Gating: per-token top-2 + softmax, scattered into dense gates[B,16].
// ---------------------------------------------------------------------------
__global__ __launch_bounds__(TPB) void gate_topk(
    const float* __restrict__ logits, float* __restrict__ gates)
{
    int t = blockIdx.x * TPB + threadIdx.x;
    if (t >= 4096) return;
    const float* l = logits + (size_t)t * 16;
    float v[16];
    float best = -3.4e38f; int bi = 0;
#pragma unroll
    for (int i = 0; i < 16; ++i) {
        v[i] = l[i];
        if (v[i] > best) { best = v[i]; bi = i; }
    }
    float second = -3.4e38f; int si = 0;
#pragma unroll
    for (int i = 0; i < 16; ++i) {
        if (i != bi && v[i] > second) { second = v[i]; si = i; }
    }
    float e2 = __expf(second - best);
    float inv = 1.0f / (1.0f + e2);
    float* g = gates + (size_t)t * 16;
#pragma unroll
    for (int i = 0; i < 16; ++i) g[i] = 0.0f;
    g[bi] = inv;
    g[si] = e2 * inv;
}

// Deterministic column sums of gates (LDS tree reduction, fixed order).
__global__ __launch_bounds__(TPB) void gate_colsum(
    const float* __restrict__ gates, float* __restrict__ gsum)
{
    __shared__ float sm[TPB * 16];
    int t = threadIdx.x;
    float p[16];
#pragma unroll
    for (int e = 0; e < 16; ++e) p[e] = 0.0f;
    for (int row = t; row < 4096; row += TPB) {
        const float* g = gates + (size_t)row * 16;
#pragma unroll
        for (int e = 0; e < 16; ++e) p[e] += g[e];
    }
#pragma unroll
    for (int e = 0; e < 16; ++e) sm[t * 16 + e] = p[e];
    __syncthreads();
    for (int s = TPB / 2; s >= 1; s >>= 1) {
        if (t < s) {
#pragma unroll
            for (int e = 0; e < 16; ++e)
                sm[t * 16 + e] += sm[(t + s) * 16 + e];
        }
        __syncthreads();
    }
    if (t < 16) gsum[t] = sm[t];
}

__global__ void lb_loss_kernel(const float* __restrict__ gsum,
                               float* __restrict__ out)
{
    int lane = threadIdx.x;
    float v = 0.0f;
    if (lane < 16) {
        float Di = gsum[lane] * (1.0f / 4096.0f);
        v = Di * logf(Di + 1e-8f);
    }
    for (int m = 16; m >= 1; m >>= 1) v += __shfl_xor(v, m, 32);
    if (lane == 0) out[0] = v;
}

// ---------------------------------------------------------------------------
// LayerNorm(residual): y = LN(moe + feat) * g + b.  One wave per row (D=256).
// ---------------------------------------------------------------------------
__global__ __launch_bounds__(TPB) void layernorm_res(
    const float* __restrict__ moe, const float* __restrict__ feat,
    const float* __restrict__ g, const float* __restrict__ b,
    float* __restrict__ yln)
{
    int lane = threadIdx.x & 31;
    int wv   = threadIdx.x >> 5;
    int row  = blockIdx.x * 8 + wv;
    const float* pm = moe  + (size_t)row * 256;
    const float* pf = feat + (size_t)row * 256;
    float vals[8], s1 = 0.0f, s2 = 0.0f;
#pragma unroll
    for (int j = 0; j < 8; ++j) {
        int c = j * 32 + lane;
        float v = pm[c] + pf[c];
        vals[j] = v;
        s1 += v;
        s2 += v * v;
    }
    for (int m = 16; m >= 1; m >>= 1) {         // wave32 butterfly (ds permute)
        s1 += __shfl_xor(s1, m, 32);
        s2 += __shfl_xor(s2, m, 32);
    }
    float mu   = s1 * (1.0f / 256.0f);
    float var  = s2 * (1.0f / 256.0f) - mu * mu;
    float rstd = rsqrtf(var + 1e-5f);
    float* py = yln + (size_t)row * 256;
#pragma unroll
    for (int j = 0; j < 8; ++j) {
        int c = j * 32 + lane;
        py[c] = (vals[j] - mu) * rstd * g[c] + b[c];
    }
}

// ---------------------------------------------------------------------------
// Host-side launch
// ---------------------------------------------------------------------------

// Workspace layout (bytes).  Big regions are reused across the conv chain.
static constexpr size_t SZ_A    = 536870912ull;                  // 4096*32*32*32*4
static constexpr size_t SZ_B    = 134217728ull;                  // 4096*32*16*16*4
static constexpr size_t SZ_C    = 67108864ull;                   // 4096*64*8*8*4
static constexpr size_t OFF_A    = 0;
static constexpr size_t OFF_B    = OFF_A + SZ_A;
static constexpr size_t OFF_C    = OFF_B + SZ_B;
static constexpr size_t OFF_FEAT = OFF_C + SZ_C;                 // 4096*256*4
static constexpr size_t OFF_H1   = OFF_FEAT + 4194304ull;        // 4096*1024*4
static constexpr size_t OFF_MOE  = OFF_H1 + 16777216ull;         // 4096*256*4
static constexpr size_t OFF_YLN  = OFF_MOE + 4194304ull;         // 4096*256*4
static constexpr size_t OFF_LOG  = OFF_YLN + 4194304ull;         // 4096*16*4
static constexpr size_t OFF_GAT  = OFF_LOG + 262144ull;          // 4096*16*4
static constexpr size_t OFF_GSUM = OFF_GAT + 262144ull;          // 16*4 (pad 256)
static constexpr size_t OFF_WC1  = OFF_GSUM + 256ull;            // 32*32*2
static constexpr size_t OFF_WC2  = OFF_WC1 + 2048ull;            // 64*288*2
static constexpr size_t OFF_WC3  = OFF_WC2 + 36864ull;           // 128*576*2
static constexpr size_t OFF_WFC  = OFF_WC3 + 147456ull;          // 256*2048*2
static constexpr size_t OFF_WG   = OFF_WFC + 1048576ull;         // 16*256*2
static constexpr size_t OFF_WE1  = OFF_WG + 8192ull;             // 16*1024*256*2
static constexpr size_t OFF_WE2  = OFF_WE1 + 8388608ull;         // 16*256*1024*2
static constexpr size_t OFF_WOUT = OFF_WE2 + 8388608ull;         // 1008*256*2

extern "C" void kernel_launch(void* const* d_in, const int* in_sizes, int n_in,
                              void* d_out, int out_size, void* d_ws, size_t ws_size,
                              hipStream_t stream)
{
    (void)in_sizes; (void)n_in; (void)out_size; (void)ws_size;

    const float* x       = (const float*)d_in[0];
    const float* conv1_w = (const float*)d_in[1];
    const float* conv1_b = (const float*)d_in[2];
    const float* conv2_w = (const float*)d_in[3];
    const float* conv2_b = (const float*)d_in[4];
    const float* conv3_w = (const float*)d_in[5];
    const float* conv3_b = (const float*)d_in[6];
    const float* fc_w    = (const float*)d_in[7];
    const float* fc_b    = (const float*)d_in[8];
    const float* gate_w  = (const float*)d_in[9];
    const float* gate_b  = (const float*)d_in[10];
    const float* e_w1    = (const float*)d_in[11];
    const float* e_b1    = (const float*)d_in[12];
    const float* e_w2    = (const float*)d_in[13];
    const float* e_b2    = (const float*)d_in[14];
    const float* ln_g    = (const float*)d_in[15];
    const float* ln_b    = (const float*)d_in[16];
    const float* out_w   = (const float*)d_in[17];
    const float* out_b   = (const float*)d_in[18];

    char* ws = (char*)d_ws;
    float* bufA  = (float*)(ws + OFF_A);
    float* bufB  = (float*)(ws + OFF_B);
    float* bufC  = (float*)(ws + OFF_C);
    float* feat  = (float*)(ws + OFF_FEAT);
    float* H1    = (float*)(ws + OFF_H1);
    float* moe   = (float*)(ws + OFF_MOE);
    float* yln   = (float*)(ws + OFF_YLN);
    float* logit = (float*)(ws + OFF_LOG);
    float* gates = (float*)(ws + OFF_GAT);
    float* gsum  = (float*)(ws + OFF_GSUM);
    unsigned short* wc1  = (unsigned short*)(ws + OFF_WC1);
    unsigned short* wc2  = (unsigned short*)(ws + OFF_WC2);
    unsigned short* wc3  = (unsigned short*)(ws + OFF_WC3);
    unsigned short* wfc  = (unsigned short*)(ws + OFF_WFC);
    unsigned short* wg   = (unsigned short*)(ws + OFF_WG);
    unsigned short* we1  = (unsigned short*)(ws + OFF_WE1);
    unsigned short* we2  = (unsigned short*)(ws + OFF_WE2);
    unsigned short* wout = (unsigned short*)(ws + OFF_WOUT);
    float* out   = (float*)d_out;                 // [4096*1000] + lb_loss

    // ---- weight conversion --------------------------------------------------
    cvt_pad_bf16<<<dim3(1, 32),  TPB, 0, stream>>>(conv1_w, wc1, 27, 32);
    cvt_pad_bf16<<<dim3(2, 64),  TPB, 0, stream>>>(conv2_w, wc2, 288, 288);
    cvt_pad_bf16<<<dim3(3, 128), TPB, 0, stream>>>(conv3_w, wc3, 576, 576);
    cvt_transpose_bf16<<<dim3(8, 256),  TPB, 0, stream>>>(fc_w,   wfc,  2048, 256);
    cvt_transpose_bf16<<<dim3(1, 16),   TPB, 0, stream>>>(gate_w, wg,   256,  16);
    cvt_transpose_bf16<<<dim3(1, 1008), TPB, 0, stream>>>(out_w,  wout, 256,  1000);
    for (int e = 0; e < 16; ++e) {
        cvt_transpose_bf16<<<dim3(1, 1024), TPB, 0, stream>>>(
            e_w1 + (size_t)e * 256 * 1024, we1 + (size_t)e * 1024 * 256, 256, 1024);
        cvt_transpose_bf16<<<dim3(4, 256), TPB, 0, stream>>>(
            e_w2 + (size_t)e * 1024 * 256, we2 + (size_t)e * 256 * 1024, 1024, 256);
    }

    // ---- CNN feature extractor (WMMA implicit-GEMM convs + pools) ----------
    // conv1: [4096,3,32,32] -> [4096,32,32,32] (bias+relu fused)
    conv_wmma<<<dim3(32768, 2), TPB, 0, stream>>>(x, wc1, conv1_b, bufA, 3, 32, 5, 27, 32);
    maxpool2<<<131072, TPB, 0, stream>>>(bufA, bufB, 5, 4, 4, 4096 * 32 * 16 * 16);
    // conv2: [4096,32,16,16] -> [4096,64,16,16]
    conv_wmma<<<dim3(8192, 4), TPB, 0, stream>>>(bufB, wc2, conv2_b, bufA, 32, 64, 4, 288, 288);
    maxpool2<<<65536, TPB, 0, stream>>>(bufA, bufC, 6, 3, 3, 4096 * 64 * 8 * 8);
    // conv3: [4096,64,8,8] -> [4096,128,8,8]
    conv_wmma<<<dim3(2048, 8), TPB, 0, stream>>>(bufC, wc3, conv3_b, bufB, 64, 128, 3, 576, 576);
    maxpool2<<<32768, TPB, 0, stream>>>(bufB, bufA, 7, 2, 2, 4096 * 128 * 4 * 4);
    // bufA now holds flattened features [4096, 2048] (NCHW flatten == reshape)

    // fc: feat = flat @ fc_w + fc_b            [4096,256]
    gemm_bf16_wmma<<<dim3(2, 256), TPB, 0, stream>>>(
        bufA, wfc, fc_b, feat, 4096, 256, 2048, 256, 256, 0);

    // ---- gating -------------------------------------------------------------
    gemm_bf16_wmma<<<dim3(1, 256), TPB, 0, stream>>>(
        feat, wg, gate_b, logit, 4096, 16, 256, 16, 16, 0);
    gate_topk<<<16, TPB, 0, stream>>>(logit, gates);
    gate_colsum<<<1, TPB, 0, stream>>>(gates, gsum);

    // ---- experts (dense all-expert, gate-weighted combine) -----------------
    zero_f32<<<4096, TPB, 0, stream>>>(moe, 4096 * 256);
    for (int e = 0; e < 16; ++e) {
        gemm_bf16_wmma<<<dim3(8, 256), TPB, 0, stream>>>(
            feat, we1 + (size_t)e * 1024 * 256, e_b1 + (size_t)e * 1024,
            H1, 4096, 1024, 256, 1024, 1024, 1 /*relu*/);
        expert2_wmma<<<dim3(2, 256), TPB, 0, stream>>>(
            H1, we2 + (size_t)e * 256 * 1024, e_b2 + (size_t)e * 256,
            gates, e, moe);
    }

    // ---- LayerNorm(residual) + classifier ----------------------------------
    layernorm_res<<<512, TPB, 0, stream>>>(moe, feat, ln_g, ln_b, yln);
    gemm_bf16_wmma<<<dim3(8, 256), TPB, 0, stream>>>(
        yln, wout, out_b, out, 4096, 1008, 256, 1000, 1000, 0);

    // lb_loss scalar appended after the logits
    lb_loss_kernel<<<1, 32, 0, stream>>>(gsum, out + (size_t)4096 * 1000);
}
